// MBartExpertLayer_12446815223983
// MI455X (gfx1250) — matrix operand order repaired and checked by reading, not compiled, not run on hardware
//
#include <hip/hip_runtime.h>
#include <hip/hip_bf16.h>
#include <math.h>

// ---------------------------------------------------------------------------
// MBart expert layer (single routed expert per batch row), fused SwiGLU MLP:
//   out[b] = ( gelu_exact(X W1[e]) * (X W3[e]) ) W2[e] * rw      (rw == 1)
// E=8, B=8, S=2048, D=1024, F=4096, fp32 in/out, bf16 WMMA compute.
// Weight slabs staged via the Tensor Data Mover (tensor_load_to_lds) into
// fp32 LDS buffers (double-buffered for W1/W3), converted to bf16 in LDS,
// consumed by v_wmma_f32_16x16x32_bf16.
// ---------------------------------------------------------------------------

typedef __attribute__((ext_vector_type(16))) __bf16 v16bf;
typedef __attribute__((ext_vector_type(8)))  float  v8f;
typedef __attribute__((ext_vector_type(4)))  unsigned int uint4v;
typedef __attribute__((ext_vector_type(4)))  unsigned int u32x4;
typedef __attribute__((ext_vector_type(8)))  int i32x8;
typedef __attribute__((ext_vector_type(4)))  int i32x4;

union FragU { uint4v q[2]; v16bf b; };

static constexpr int Bv = 8, Sv = 2048, Dv = 1024, Fv = 4096;
static constexpr int ROWS = 64;   // seq rows per workgroup
static constexpr int DB   = 256;  // output d-columns per workgroup
static constexpr int FB   = 32;   // F chunk (inner dim of 2nd GEMM)
static constexpr int KB   = 128;  // K slab staged per step of 1st GEMM

// LDS layout (bytes). bf16 tile rows padded to 16B multiples for aligned
// ds_load_b128 and bank spreading.
static constexpr int XS_STRIDE  = Dv * 2;                    // 2048 B
static constexpr int W1T_OFF    = ROWS * XS_STRIDE;          // 131072
static constexpr int WT_STRIDE  = KB * 2 + 16;               // 272 B
static constexpr int W3T_OFF    = W1T_OFF + FB * WT_STRIDE;  // +8704
static constexpr int W2T_OFF    = W3T_OFF + FB * WT_STRIDE;  // +8704
static constexpr int W2T_STRIDE = FB * 2 + 16;               // 80 B
static constexpr int HS_OFF     = W2T_OFF + DB * W2T_STRIDE; // +20480
static constexpr int HS_STRIDE  = FB * 2 + 16;               // 80 B
// fp32 TDM staging: two (W1 slab + W3 slab) buffers + one W2 slab buffer.
static constexpr int WSLAB_B    = KB * FB * 4;               // 16384 B
static constexpr int ST_W_OFF   = HS_OFF + ROWS * HS_STRIDE; // 174080
static constexpr int ST_W2_OFF  = ST_W_OFF + 2 * 2 * WSLAB_B;// +65536 = 239616
static constexpr int W2SLAB_B   = FB * DB * 4;               // 32768 B
static constexpr int SMEM_BYTES = ST_W2_OFF + W2SLAB_B;      // 272384 (<320K)

__device__ __forceinline__ unsigned int pack_bf16(float a, float b) {
  unsigned int ua = __float_as_uint(a);
  unsigned int ub = __float_as_uint(b);
  ua += 0x7FFFu + ((ua >> 16) & 1u);   // round-to-nearest-even
  ub += 0x7FFFu + ((ub >> 16) & 1u);
  return (ua >> 16) | (ub & 0xFFFF0000u);
}
__device__ __forceinline__ unsigned short f2bf(float a) {
  unsigned int u = __float_as_uint(a);
  u += 0x7FFFu + ((u >> 16) & 1u);
  return (unsigned short)(u >> 16);
}

// ---- Tensor Data Mover: 2D fp32 tile (tile_w x tile_h, row stride in elems)
// from global into LDS at byte offset lds_off. D# packed per CDNA5 ISA ch.8.
__device__ __forceinline__ void tdm_load_2d_f32(unsigned lds_off,
                                                const float* gptr,
                                                unsigned tile_w, unsigned tile_h,
                                                unsigned row_stride_elems) {
  unsigned long long ga = (unsigned long long)(size_t)gptr;
  // group0: [1:0]=count=1 | [63:32]=lds_addr | [120:64]=global_addr | [127:126]=type 2
  u32x4 g0 = { 1u,
               lds_off,
               (unsigned)(ga & 0xFFFFFFFFu),
               (unsigned)((ga >> 32) & 0x01FFFFFFu) | 0x80000000u };
  // group1: data_size=4B; tensor dims == tile dims (tile fully interior);
  // tile_dim0=tile_w, tile_dim1=tile_h; tensor_dim0_stride=row stride.
  i32x8 g1 = { (int)(2u << 16),                                  // data_size=2 (4B)
               (int)((tile_w & 0xFFFFu) << 16),                  // tensor_dim0 lo16
               (int)((tile_w >> 16) | ((tile_h & 0xFFFFu) << 16)),// dim0 hi | dim1 lo
               (int)((tile_h >> 16) | (tile_w << 16)),           // dim1 hi | tile_dim0
               (int)(tile_h & 0xFFFFu),                          // tile_dim1 (tile_dim2=0)
               (int)row_stride_elems,                            // dim0_stride lo32
               0, 0 };
  i32x4 zz = { 0, 0, 0, 0 };
#if defined(__clang_major__) && (__clang_major__ >= 23)
  i32x8 z8 = { 0, 0, 0, 0, 0, 0, 0, 0 };
  __builtin_amdgcn_tensor_load_to_lds(g0, g1, zz, zz, z8, 0);
#else
  __builtin_amdgcn_tensor_load_to_lds(g0, g1, zz, zz, 0);
#endif
}

// A-matrix fragment (16x32 bf16) from row-major [m][k] LDS tile.
// ISA layout: lanes 0-15 -> M=lane, K = k0+{0..7, 16..23};
//             lanes 16-31 -> M=lane-16, K = k0+{8..15, 24..31}.
__device__ __forceinline__ v16bf load_fragA(const char* base, int stride,
                                            int m0, int k0, int lane) {
  const int l = lane & 15, hi = lane >> 4;
  const char* p = base + (m0 + l) * stride + (k0 + hi * 8) * 2;
  FragU u;
  u.q[0] = *(const uint4v*)p;
  u.q[1] = *(const uint4v*)(p + 32);
  return u.b;
}

// B-matrix fragment (32x16 bf16, K x N) from [n][k]-transposed LDS tile.
// ISA layout: lanes 0-15 -> N=lane, K = k0+{0..7, 16..23}; lanes 16-31 -> N, K = k0+{8..15, 24..31}.
__device__ __forceinline__ v16bf load_fragB(const char* base, int stride,
                                            int n0, int k0, int lane) {
  const int l = lane & 15, hi = lane >> 4;
  const char* p = base + (n0 + l) * stride + (k0 + hi * 8) * 2;
  FragU u;
  u.q[0] = *(const uint4v*)p;
  u.q[1] = *(const uint4v*)(p + 32);
  return u.b;
}

__global__ void __launch_bounds__(256)
moe_expert_kernel(const float* __restrict__ x,
                  const float* __restrict__ w1,
                  const float* __restrict__ w2,
                  const float* __restrict__ w3,
                  const int* __restrict__ langs,
                  float* __restrict__ out) {
  extern __shared__ char smem[];
  char* Xs  = smem;            // 64 x 1024 bf16
  char* W1t = smem + W1T_OFF;  // 32(f) x 128(k) bf16, transposed
  char* W3t = smem + W3T_OFF;
  char* W2t = smem + W2T_OFF;  // 256(d) x 32(f) bf16, transposed
  char* Hs  = smem + HS_OFF;   // 64 x 32 bf16

  const int dChunk = blockIdx.x;          // 0..3
  const int sTile  = blockIdx.y;          // 0..31
  const int b      = blockIdx.z;          // 0..7
  const int tid    = threadIdx.x;
  const int lane   = tid & 31;
  const int wv     = tid >> 5;            // wave 0..7
  const int r      = wv & 3;              // 16-row tile within 64 rows
  const int cg     = wv >> 2;             // phase1 col tile / phase2 col group
  const int s0     = sTile * ROWS;
  const int d0     = dChunk * DB;

  // Routing: exactly one lang per row, codes 4..11 -> experts 0..7; rw == 1.
  int e = langs[b] - 4;
  e = e < 0 ? 0 : (e > 7 ? 7 : e);
  const float rw = 1.0f;

  const float* xrow = x  + ((size_t)b * Sv + s0) * Dv;
  const float* W1g  = w1 + (size_t)e * Dv * Fv;
  const float* W3g  = w3 + (size_t)e * Dv * Fv;
  const float* W2g  = w2 + (size_t)e * Fv * Dv;

  // ---- Stage X tile once: 64x1024 fp32 -> bf16 in LDS (contiguous) ----
  for (int i = tid; i < ROWS * Dv / 2; i += 256) {
    const float2 v = ((const float2*)xrow)[i];
    ((unsigned int*)Xs)[i] = pack_bf16(v.x, v.y);
  }
  __syncthreads();

  const v8f vzero = {0.f, 0.f, 0.f, 0.f, 0.f, 0.f, 0.f, 0.f};
  v8f acc[8];
#pragma unroll
  for (int j = 0; j < 8; ++j) acc[j] = vzero;

  for (int f0 = 0; f0 < Fv; f0 += FB) {
    // Prologue: wave0 DMAs the first W1/W3 K-slab into staging buffer 0;
    // wave1 DMAs the W2 chunk (overlaps all of phase 1). Per-wave TENSORcnt.
    if (wv == 0) {
      tdm_load_2d_f32(ST_W_OFF,           W1g + (size_t)0 * Fv + f0, FB, KB, Fv);
      tdm_load_2d_f32(ST_W_OFF + WSLAB_B, W3g + (size_t)0 * Fv + f0, FB, KB, Fv);
    }
    if (wv == 1) {
      tdm_load_2d_f32(ST_W2_OFF, W2g + (size_t)f0 * Dv + d0, DB, FB, Dv);
    }

    v8f accA = vzero, accB = vzero;

    // ---- Phase 1: a = X W1 chunk, bgate = X W3 chunk (K = 1024) ----
    for (int kb = 0; kb < Dv; kb += KB) {
      const int p = (kb >> 7) & 1;                 // staging buffer parity
      const char* stg = smem + ST_W_OFF + p * (2 * WSLAB_B);

      if (wv == 0) __builtin_amdgcn_s_wait_tensorcnt(0);  // slab(kb) landed
      __syncthreads();
      if (wv == 0 && kb + KB < Dv) {               // DMA next slab, other buffer
        const unsigned np = ST_W_OFF + (p ^ 1) * (2 * WSLAB_B);
        tdm_load_2d_f32(np,           W1g + (size_t)(kb + KB) * Fv + f0, FB, KB, Fv);
        tdm_load_2d_f32(np + WSLAB_B, W3g + (size_t)(kb + KB) * Fv + f0, FB, KB, Fv);
      }

      // convert staged fp32 slab -> transposed bf16 [f][k] tiles
      {
        const float* s1 = (const float*)stg;
        const float* s3 = (const float*)(stg + WSLAB_B);
        for (int i = tid; i < FB * KB / 2; i += 256) {
          const int f  = i & (FB - 1);
          const int kp = i >> 5;  // k pair index 0..63
          ((unsigned int*)(W1t + f * WT_STRIDE))[kp] =
              pack_bf16(s1[(2 * kp) * FB + f], s1[(2 * kp + 1) * FB + f]);
          ((unsigned int*)(W3t + f * WT_STRIDE))[kp] =
              pack_bf16(s3[(2 * kp) * FB + f], s3[(2 * kp + 1) * FB + f]);
        }
      }
      __syncthreads();
#pragma unroll
      for (int kk = 0; kk < KB; kk += 32) {
        const v16bf af = load_fragA(Xs, XS_STRIDE, r * 16, kb + kk, lane);
        const v16bf b1 = load_fragB(W1t, WT_STRIDE, cg * 16, kk, lane);
        const v16bf b3 = load_fragB(W3t, WT_STRIDE, cg * 16, kk, lane);
        accA = __builtin_amdgcn_wmma_f32_16x16x32_bf16(false, af, false, b1,
                                                       (short)0, accA, false, false);
        accB = __builtin_amdgcn_wmma_f32_16x16x32_bf16(false, af, false, b3,
                                                       (short)0, accB, false, false);
      }
      __syncthreads();
    }

    // ---- h = gelu_exact(a) * bgate  -> Hs (bf16) ----
    {
      const int l = lane & 15, hi = lane >> 4;
      const int n = cg * 16 + l;
#pragma unroll
      for (int i = 0; i < 8; ++i) {
        const int m = r * 16 + hi * 8 + i;
        const float a = accA[i];
        const float g = 0.5f * a * (1.0f + erff(a * 0.70710678118654752f));
        ((unsigned short*)(Hs + m * HS_STRIDE))[n] = f2bf(g * accB[i]);
      }
    }

    // ---- W2 chunk: wait for TDM, convert to transposed bf16 [d][f] ----
    if (wv == 1) __builtin_amdgcn_s_wait_tensorcnt(0);
    __syncthreads();
    {
      const float* s2 = (const float*)(smem + ST_W2_OFF);
      for (int i = tid; i < DB * FB / 2; i += 256) {
        const int d  = i & (DB - 1);
        const int fp = i >> 8;  // f pair index 0..15
        ((unsigned int*)(W2t + d * W2T_STRIDE))[fp] =
            pack_bf16(s2[(2 * fp) * DB + d], s2[(2 * fp + 1) * DB + d]);
      }
    }
    __syncthreads();

    // ---- Phase 2: out_acc += h(64x32) @ W2chunk(32x256) ----
    {
      const v16bf ha = load_fragA(Hs, HS_STRIDE, r * 16, 0, lane);
#pragma unroll
      for (int j = 0; j < 8; ++j) {
        const v16bf wb = load_fragB(W2t, W2T_STRIDE, (cg * 8 + j) * 16, 0, lane);
        acc[j] = __builtin_amdgcn_wmma_f32_16x16x32_bf16(false, ha, false, wb,
                                                         (short)0, acc[j], false, false);
      }
    }
    __syncthreads();  // protect Hs/W1t/W2t/staging before next chunk
  }

  // ---- write out (fp32), fully covering the [64 x 256] tile ----
  {
    const int l = lane & 15, hi = lane >> 4;
    float* orow = out + ((size_t)b * Sv + s0) * Dv + d0;
#pragma unroll
    for (int j = 0; j < 8; ++j) {
      const int n = (cg * 8 + j) * 16 + l;
#pragma unroll
      for (int i = 0; i < 8; ++i) {
        const int m = r * 16 + hi * 8 + i;
        orow[(size_t)m * Dv + n] = acc[j][i] * rw;
      }
    }
  }
}

extern "C" void kernel_launch(void* const* d_in, const int* in_sizes, int n_in,
                              void* d_out, int out_size, void* d_ws, size_t ws_size,
                              hipStream_t stream) {
  const float* x    = (const float*)d_in[0];  // hidden_states [B,S,D]
  const float* w1   = (const float*)d_in[1];  // [E,D,F]
  const float* w2   = (const float*)d_in[2];  // [E,F,D]
  const float* w3   = (const float*)d_in[3];  // [E,D,F]
  const int*   lang = (const int*)d_in[4];    // [B,1]
  float* out = (float*)d_out;

  dim3 grid(Dv / DB, Sv / ROWS, Bv);  // 4 x 32 x 8 = 1024 workgroups
  dim3 block(256);                    // 8 wave32s
  hipLaunchKernelGGL(moe_expert_kernel, grid, block, SMEM_BYTES, stream,
                     x, w1, w2, w3, lang, out);
}
